// GPM_82927228551563
// MI455X (gfx1250) — compile-verified, matrix operand order checked by reading, not run on hardware
//
#include <hip/hip_runtime.h>
#include <hip/hip_bf16.h>

typedef __bf16 bf16;
typedef __attribute__((ext_vector_type(16))) __bf16 v16bf;
typedef __attribute__((ext_vector_type(8)))  __bf16 v8bf;
typedef __attribute__((ext_vector_type(8)))  float  v8f;
typedef unsigned int u32x4 __attribute__((ext_vector_type(4)));
typedef int          i32x8 __attribute__((ext_vector_type(8)));
typedef int          i32x4 __attribute__((ext_vector_type(4)));

#define CDIM 512
#define TOPK 5
#define NWAVES 4
#define BLOCK_MAIN (NWAVES * 32)
#define ROWS_PER_BLOCK 16
#define BPITCH 520                    // bf16 elems per LDS B row (1024B + 16B TDM pad)
#define BUFBYTES (16 * BPITCH * 2)    // one 16-row B tile in LDS = 16640 B
#define CANDS (NWAVES * 16 * TOPK)    // 320 packed candidates per query row
#define IDXMASK 0x7FFFu               // 15 bits: M = 32768 memory rows

static __device__ __forceinline__ unsigned short f2bf(float f) {
    unsigned u = __float_as_uint(f);
    unsigned r = u + 0x7FFFu + ((u >> 16) & 1u);   // round-to-nearest-even
    return (unsigned short)(r >> 16);
}

// Pack 15-bit column index into low mantissa bits of fp32 score (order-preserving).
static __device__ __forceinline__ float packsi(float s, int col) {
    return __uint_as_float((__float_as_uint(s) & ~IDXMASK) | (unsigned)col);
}

// Sorted-descending top-5 bubble insert on packed scores (branchless min/max chain).
static __device__ __forceinline__ void ins5(float (&t)[TOPK], float s) {
#pragma unroll
    for (int k = 0; k < TOPK; ++k) {
        float mx = fmaxf(s, t[k]);
        float mn = fminf(s, t[k]);
        t[k] = mx;
        s = mn;
    }
}

// One TDM descriptor: copy a 16 x 512 bf16 tile (rows contiguous, stride 512)
// from global to LDS, padding 4 DWORDs after every 256 DWORDs (row) -> pitch 520.
static __device__ __forceinline__ void tdm_stage_tile(unsigned lds_byte, const bf16* gsrc) {
    unsigned long long ga = (unsigned long long)(size_t)gsrc;
    u32x4 g0;
    g0[0] = 1u;                                            // count=1, user mode
    g0[1] = lds_byte;                                      // LDS dest (bytes)
    g0[2] = (unsigned)(ga & 0xFFFFFFFFu);                  // global addr lo
    g0[3] = (unsigned)((ga >> 32) & 0x01FFFFFFu) | (2u << 30); // addr[56:32] | type=2
    i32x8 g1;
    g1[0] = (int)((1u << 16)      // data_size = 2 bytes
                | (1u << 20)      // pad_enable
                | (7u << 22)      // pad_interval: 256 DWORDs
                | (3u << 25));    // pad_amount: 4 DWORDs (16 B)
    g1[1] = (int)(512u << 16);    // tensor_dim0 = 512 (low 16)
    g1[2] = (int)(0x8000u << 16); // tensor_dim0 hi=0 | tensor_dim1 = 32768 (low 16)
    g1[3] = (int)(512u << 16);    // tensor_dim1 hi=0 | tile_dim0 = 512
    g1[4] = 16;                   // tile_dim1 = 16 | tile_dim2 = 0
    g1[5] = 512;                  // tensor_dim0_stride = 512 (low 32)
    g1[6] = 0;
    g1[7] = 0;
    i32x4 z4 = {0, 0, 0, 0};
    i32x8 z8 = {0, 0, 0, 0, 0, 0, 0, 0};
    __builtin_amdgcn_tensor_load_to_lds(g0, g1, z4, z4, z8, 0);
}

// One wave per row: L2-normalize a 512-wide fp32 row, optional quality fold, emit bf16.
__global__ void l2norm_bf16_rows(const float* __restrict__ src,
                                 const float* __restrict__ quality,
                                 bf16* __restrict__ dst, int rows) {
    int wv   = (blockIdx.x * blockDim.x + threadIdx.x) >> 5;
    int lane = threadIdx.x & 31;
    if (wv >= rows) return;
    const float4* p = (const float4*)(src + (size_t)wv * CDIM);
    float4 v[4];
    float ss = 0.f;
#pragma unroll
    for (int i = 0; i < 4; ++i) {
        v[i] = p[lane + i * 32];
        ss += v[i].x * v[i].x + v[i].y * v[i].y + v[i].z * v[i].z + v[i].w * v[i].w;
    }
#pragma unroll
    for (int off = 16; off >= 1; off >>= 1) ss += __shfl_xor(ss, off, 32);
    float sc = 1.f / fmaxf(sqrtf(ss), 1e-12f);
    if (quality) sc *= quality[wv];
    bf16* drow = dst + (size_t)wv * CDIM;
#pragma unroll
    for (int i = 0; i < 4; ++i) {
        ushort4 pk;
        pk.x = f2bf(v[i].x * sc); pk.y = f2bf(v[i].y * sc);
        pk.z = f2bf(v[i].z * sc); pk.w = f2bf(v[i].w * sc);
        *(ushort4*)(drow + (size_t)(lane + i * 32) * 4) = pk;
    }
}

// Fused: WMMA similarity (TDM double-buffered B) + streaming top-5 + softmax + gather.
__global__ __launch_bounds__(BLOCK_MAIN)
void retrieve_topk_wmma(const float* __restrict__ x,
                        const float* __restrict__ mem,    // raw fp32 memory_mean [M,C]
                        const bf16*  __restrict__ qn,     // normalized queries  [N,C] bf16
                        const bf16*  __restrict__ mnq,    // normalized*quality  [M,C] bf16
                        float* __restrict__ out, int M) {
    extern __shared__ __align__(16) char sB[];            // NWAVES * 2 * BUFBYTES
    __shared__ float sCs[ROWS_PER_BLOCK][CANDS];          // packed score|idx
    __shared__ float sW[ROWS_PER_BLOCK][TOPK];
    __shared__ int   sI[ROWS_PER_BLOCK][TOPK];

    const int t    = threadIdx.x;
    const int lane = t & 31;
    const int rlo  = lane & 15;
    const int koff = (lane < 16) ? 0 : 8;                 // bf16 fragment K sub-offset
    const int swv  = __builtin_amdgcn_readfirstlane((int)(t >> 5));
    const size_t qrow0 = (size_t)blockIdx.x * ROWS_PER_BLOCK;

    // ---- A fragments: loop-invariant, load straight into registers (16 x v16bf) ----
    v16bf afrag[16];
    {
        const bf16* aBase = qn + (qrow0 + rlo) * CDIM;
#pragma unroll
        for (int ks = 0; ks < 16; ++ks) {
            v8bf lo = *(const v8bf*)(aBase + ks * 32 + koff);
            v8bf hi = *(const v8bf*)(aBase + ks * 32 + 16 + koff);
#pragma unroll
            for (int j = 0; j < 8; ++j) { afrag[ks][j] = lo[j]; afrag[ks][j + 8] = hi[j]; }
        }
    }

    float ts[8][TOPK];
#pragma unroll
    for (int i = 0; i < 8; ++i)
#pragma unroll
        for (int k = 0; k < TOPK; ++k) ts[i][k] = -3.4e38f;

    const int ntiles = M >> 4;
    const unsigned wvLds = __builtin_amdgcn_readfirstlane(
        (int)((unsigned)(size_t)sB + (unsigned)swv * 2u * BUFBYTES));
    const char* wvBuf = sB + (size_t)swv * 2u * BUFBYTES;

    // Prime the pipeline: stage this wave's first B tile.
    tdm_stage_tile(wvLds, mnq + (size_t)swv * 16 * CDIM);

    int cur = 0;
    for (int mt = swv; mt < ntiles; mt += NWAVES) {
        const int nxt = mt + NWAVES;
        if (nxt < ntiles) {
            tdm_stage_tile(wvLds + (unsigned)((cur ^ 1) * BUFBYTES),
                           mnq + (size_t)nxt * 16 * CDIM);
            __builtin_amdgcn_s_wait_tensorcnt(1);   // current tile landed in LDS
        } else {
            __builtin_amdgcn_s_wait_tensorcnt(0);
        }

        const bf16* bRow = (const bf16*)(wvBuf + cur * BUFBYTES) + rlo * BPITCH;
        v8f c = {0.f, 0.f, 0.f, 0.f, 0.f, 0.f, 0.f, 0.f};

        // K = 512 = 16 x 32, with B fragments register double-buffered so each
        // step's ds_loads issue one WMMA ahead of their consumer (hides LDS latency).
        v8bf blo[2], bhi[2];
        blo[0] = *(const v8bf*)(bRow + koff);
        bhi[0] = *(const v8bf*)(bRow + 16 + koff);
#pragma unroll
        for (int ks = 0; ks < 16; ++ks) {
            const int cb = ks & 1, nb = cb ^ 1;
            if (ks < 15) {
                blo[nb] = *(const v8bf*)(bRow + (ks + 1) * 32 + koff);
                bhi[nb] = *(const v8bf*)(bRow + (ks + 1) * 32 + 16 + koff);
            }
            v16bf b;
#pragma unroll
            for (int j = 0; j < 8; ++j) { b[j] = blo[cb][j]; b[j + 8] = bhi[cb][j]; }
            c = __builtin_amdgcn_wmma_f32_16x16x32_bf16(false, afrag[ks], false, b,
                                                        (short)0, c, false, false);
        }

        const int col = (mt << 4) + rlo;            // this lane's memory-row index
#pragma unroll
        for (int i = 0; i < 8; ++i) {
            float sp = packsi(c[i], col);
            if (sp > ts[i][TOPK - 1]) ins5(ts[i], sp);
        }
        cur ^= 1;
    }

    // Dump per-lane candidates; lanes/waves cover disjoint column subsets -> merge is exact.
    {
        const int rowhi = (lane < 16) ? 0 : 8;
        const int base  = swv * (16 * TOPK) + rlo * TOPK;
#pragma unroll
        for (int i = 0; i < 8; ++i) {
            const int row = i + rowhi;
#pragma unroll
            for (int k = 0; k < TOPK; ++k) sCs[row][base + k] = ts[i][k];
        }
    }
    __syncthreads();

    // 16 threads: merge 320 candidates per row, then softmax over top-5.
    if (t < ROWS_PER_BLOCK) {
        float bs[TOPK];
#pragma unroll
        for (int k = 0; k < TOPK; ++k) bs[k] = -3.4e38f;
        for (int j = 0; j < CANDS; ++j) {
            float s = sCs[t][j];
            if (s > bs[TOPK - 1]) ins5(bs, s);
        }
        float m0 = bs[0], e[TOPK], sum = 0.f;
#pragma unroll
        for (int k = 0; k < TOPK; ++k) { e[k] = expf(bs[k] - m0); sum += e[k]; }
        float inv = 1.f / sum;
#pragma unroll
        for (int k = 0; k < TOPK; ++k) {
            sW[t][k] = e[k] * inv;
            sI[t][k] = (int)(__float_as_uint(bs[k]) & IDXMASK);
        }
    }
    __syncthreads();

    // out[row] = x[row] + 0.5 * sum_k w_k * mem[idx_k]; 8 threads x 64 cols per row.
    {
        const int row = t >> 3;
        const int seg = t & 7;
        const size_t g = (qrow0 + row) * CDIM + (size_t)seg * 64;
        const float4* xp = (const float4*)(x + g);
        float4* op = (float4*)(out + g);
        float  w[TOPK]; size_t off[TOPK];
#pragma unroll
        for (int k = 0; k < TOPK; ++k) {
            w[k]   = sW[row][k];
            off[k] = (size_t)sI[row][k] * CDIM + (size_t)seg * 64;
        }
#pragma unroll 4
        for (int j = 0; j < 16; ++j) {
            float4 xv = xp[j];
            float ax = 0.f, ay = 0.f, az = 0.f, aw = 0.f;
#pragma unroll
            for (int k = 0; k < TOPK; ++k) {
                float4 mv = *(const float4*)(mem + off[k] + (size_t)j * 4);
                ax += w[k] * mv.x; ay += w[k] * mv.y;
                az += w[k] * mv.z; aw += w[k] * mv.w;
            }
            float4 r;
            r.x = xv.x + 0.5f * ax; r.y = xv.y + 0.5f * ay;
            r.z = xv.z + 0.5f * az; r.w = xv.w + 0.5f * aw;
            op[j] = r;
        }
    }
}

extern "C" void kernel_launch(void* const* d_in, const int* in_sizes, int n_in,
                              void* d_out, int out_size, void* d_ws, size_t ws_size,
                              hipStream_t stream) {
    (void)n_in; (void)out_size; (void)ws_size;
    const float* x    = (const float*)d_in[0];   // [B*S, C] fp32
    const float* mm   = (const float*)d_in[1];   // [M, C]   fp32
    const float* qual = (const float*)d_in[2];   // [M]      fp32
    float* out = (float*)d_out;

    const int N = in_sizes[0] / CDIM;            // 8192
    const int M = in_sizes[2];                   // 32768

    bf16* qn  = (bf16*)d_ws;                     // N*C bf16  (8 MB)
    bf16* mnq = qn + (size_t)N * CDIM;           // M*C bf16  (32 MB)

    l2norm_bf16_rows<<<(M + 7) / 8, 256, 0, stream>>>(mm, qual, mnq, M);
    l2norm_bf16_rows<<<(N + 7) / 8, 256, 0, stream>>>(x, nullptr, qn, N);

    const size_t dynLds = (size_t)NWAVES * 2 * BUFBYTES;   // 130 KB B staging
    retrieve_topk_wmma<<<N / ROWS_PER_BLOCK, BLOCK_MAIN, dynLds, stream>>>(
        x, mm, qn, mnq, out, M);
}
